// CausalSelfAttention_69174743269902
// MI455X (gfx1250) — matrix operand order repaired and checked by reading, not compiled, not run on hardware
//
#include <hip/hip_runtime.h>

// ---------------------------------------------------------------------------
// CDNA5 (gfx1250) causal self-attention.
//  - bf16 WMMA (v_wmma_f32_16x16x32_bf16), f32 accumulation, wave32.
//  - GLOBAL_LOAD_ASYNC_TO_LDS_B128 (ASYNCcnt) for tile staging, with
//    double-buffered GEMM K-loop.
// ---------------------------------------------------------------------------

typedef __attribute__((ext_vector_type(16))) __bf16 v16bf;
typedef __attribute__((ext_vector_type(8)))  float  v8f;

struct BF16Frag { union { v16bf v; uint4 q[2]; }; };

// A-matrix fragment (16x32 bf16): lane L: row = L%16, hl = L/16,
// VGPR0..3 hold K = hl*8..hl*8+7, VGPR4..7 hold K = 16+hl*8..16+hl*8+7.
__device__ __forceinline__ void load_fragA(BF16Frag& f, const __bf16* rowPtr,
                                           int kOff, int hl) {
  f.q[0] = *(const uint4*)(rowPtr + kOff + hl * 8);
  f.q[1] = *(const uint4*)(rowPtr + kOff + 16 + hl * 8);
}

// B-matrix fragment (32x16 bf16, stored N-major [n][k]): lane L: col = L%16,
// hl = L/16, VGPR0..3 hold K = hl*8..hl*8+7, VGPR4..7 hold K = 16+hl*8..16+hl*8+7.
__device__ __forceinline__ void load_fragB(BF16Frag& f, const __bf16* rowPtr,
                                           int kOff, int hl) {
  f.q[0] = *(const uint4*)(rowPtr + kOff + hl * 8);
  f.q[1] = *(const uint4*)(rowPtr + kOff + 16 + hl * 8);
}

__device__ __forceinline__ v8f wmma_bf16(v16bf a, v16bf b, v8f c) {
  return __builtin_amdgcn_wmma_f32_16x16x32_bf16(false, a, false, b,
                                                 (short)0, c, false, false);
}

// Async DMA: 16 bytes per lane, global -> LDS, tracked by ASYNCcnt.
// VDST operand = per-lane LDS byte address (low 32 bits of generic pointer).
__device__ __forceinline__ void async_copy16(const void* g, const void* l) {
  asm volatile("global_load_async_to_lds_b128 %0, %1, off"
               :: "v"((unsigned)(unsigned long long)l), "v"(g)
               : "memory");
}
__device__ __forceinline__ void wait_async_le4() {
  asm volatile("s_wait_asynccnt 0x4" ::: "memory");
}
__device__ __forceinline__ void wait_async_0() {
  asm volatile("s_wait_asynccnt 0x0" ::: "memory");
}

// ---------------------------------------------------------------------------
// Kernel 0: fp32 -> bf16 elementwise (for x)
// ---------------------------------------------------------------------------
__global__ __launch_bounds__(256) void convert_f32_to_bf16(
    const float* __restrict__ in, void* __restrict__ outp, long n) {
  __bf16* out = (__bf16*)outp;
  const long i = ((long)blockIdx.x * 256 + threadIdx.x) * 8;
  if (i >= n) return;
  const float4* s = (const float4*)(in + i);
  const float4 a = s[0], b = s[1];
  union { __bf16 h[8]; uint4 q; } u;
  u.h[0] = (__bf16)a.x; u.h[1] = (__bf16)a.y;
  u.h[2] = (__bf16)a.z; u.h[3] = (__bf16)a.w;
  u.h[4] = (__bf16)b.x; u.h[5] = (__bf16)b.y;
  u.h[6] = (__bf16)b.z; u.h[7] = (__bf16)b.w;
  *(uint4*)(out + i) = u.q;
}

// ---------------------------------------------------------------------------
// Kernel 1: transpose fp32 [R][Cc] -> bf16 [Cc][R]   (coalesced both sides)
// ---------------------------------------------------------------------------
__global__ __launch_bounds__(256) void transpose_to_bf16(
    const float* __restrict__ W, void* __restrict__ Wtv, int R, int Cc) {
  __shared__ float tile[32][33];
  __bf16* Wt = (__bf16*)Wtv;
  const int tx = threadIdx.x & 31, ty = threadIdx.x >> 5;  // 32 x 8
  const int r0 = blockIdx.y * 32, c0 = blockIdx.x * 32;
#pragma unroll
  for (int i = 0; i < 4; ++i)
    tile[ty + i * 8][tx] = W[(size_t)(r0 + ty + i * 8) * Cc + c0 + tx];
  __syncthreads();
#pragma unroll
  for (int i = 0; i < 4; ++i)
    Wt[(size_t)(c0 + ty + i * 8) * R + r0 + tx] = (__bf16)tile[tx][ty + i * 8];
}

// ---------------------------------------------------------------------------
// Kernel 2/4: GEMM  Out[M][N] = A[M][K](bf16) * Bt[N][K]^T(bf16) + bias
//   Block tile 128x128, BK=32; 8 waves, wave tile 32x64 = 2x4 WMMA tiles.
//   Double-buffered LDS, staged with async-to-LDS DMA (4 ops/wave/tile).
// ---------------------------------------------------------------------------
template <bool OUT_F32>
__global__ __launch_bounds__(256) void gemm_bf16_wmma(
    const void* __restrict__ Ap, const void* __restrict__ Btp,
    const float* __restrict__ bias, void* __restrict__ Outp,
    int M, int N, int K) {
  __shared__ __bf16 As[2][128 * 32];
  __shared__ __bf16 Bs[2][128 * 32];
  const __bf16* A  = (const __bf16*)Ap;
  const __bf16* Bt = (const __bf16*)Btp;

  const int tid = threadIdx.x;
  const int lane = tid & 31, w = tid >> 5;
  const int wm = w & 3, wn = w >> 2;          // waves: 4 along M, 2 along N
  const int n16 = lane & 15, hl = lane >> 4;
  const int m0 = blockIdx.y * 128, n0 = blockIdx.x * 128;

  v8f acc[2][4];
#pragma unroll
  for (int i = 0; i < 2; ++i)
#pragma unroll
    for (int j = 0; j < 4; ++j)
#pragma unroll
      for (int v = 0; v < 8; ++v) acc[i][j][v] = 0.0f;

  const int srow = tid >> 1, seg = tid & 1;   // staging: 2 threads per row

  // Issue one tile's async stage: 2x16B for A + 2x16B for B per thread.
  auto stage_async = [&](int buf, int kt) {
    const __bf16* ga = A + (size_t)(m0 + srow) * K + kt + seg * 16;
    const __bf16* la = &As[buf][srow * 32 + seg * 16];
    async_copy16(ga, la);
    async_copy16(ga + 8, la + 8);
    const __bf16* gb = Bt + (size_t)(n0 + srow) * K + kt + seg * 16;
    const __bf16* lb = &Bs[buf][srow * 32 + seg * 16];
    async_copy16(gb, lb);
    async_copy16(gb + 8, lb + 8);
  };

  const int nk = K >> 5;
  stage_async(0, 0);

  for (int it = 0; it < nk; ++it) {
    const int cur = it & 1;
    if (it + 1 < nk) {
      stage_async(cur ^ 1, (it + 1) << 5);
      wait_async_le4();   // in-order completion => current tile is resident
    } else {
      wait_async_0();
    }
    __syncthreads();

    BF16Frag af[2];
    load_fragA(af[0], &As[cur][(wm * 32 + n16) * 32], 0, hl);
    load_fragA(af[1], &As[cur][(wm * 32 + 16 + n16) * 32], 0, hl);
#pragma unroll
    for (int nt = 0; nt < 4; ++nt) {
      BF16Frag bf;
      load_fragB(bf, &Bs[cur][(wn * 64 + nt * 16 + n16) * 32], 0, hl);
      acc[0][nt] = wmma_bf16(af[0].v, bf.v, acc[0][nt]);
      acc[1][nt] = wmma_bf16(af[1].v, bf.v, acc[1][nt]);
    }
    __syncthreads();  // all reads of `cur` done before it is re-staged
  }

  // ---- epilogue: C/D layout -> global ----
#pragma unroll
  for (int mt = 0; mt < 2; ++mt) {
#pragma unroll
    for (int nt = 0; nt < 4; ++nt) {
      const int c = n0 + wn * 64 + nt * 16 + n16;
      const float bv = bias[c];
#pragma unroll
      for (int v = 0; v < 8; ++v) {
        const int r = m0 + wm * 32 + mt * 16 + v + 8 * hl;
        const float val = acc[mt][nt][v] + bv;
        if constexpr (OUT_F32)
          ((float*)Outp)[(size_t)r * N + c] = val;
        else
          ((__bf16*)Outp)[(size_t)r * N + c] = (__bf16)val;
      }
    }
  }
}

// ---------------------------------------------------------------------------
// Kernel 3: flash attention (causal).  One block per (b,h) x 128 query rows.
// qkv bf16 [B*S][3072] (q|k|v, head-interleaved).  y bf16 [B*S][1024].
// ---------------------------------------------------------------------------
__global__ __launch_bounds__(256) void attn_flash_wmma(
    const void* __restrict__ qkvp, void* __restrict__ yp, int S) {
  extern __shared__ __align__(16) char smem[];
  __bf16* Qs  = (__bf16*)smem;        // [128][64]
  __bf16* Ks  = Qs + 128 * 64;        // [128][64]
  __bf16* Vts = Ks + 128 * 64;        // [64][128]  (transposed V)
  __bf16* Ps  = Vts + 64 * 128;       // 8 waves x [16][128]

  const __bf16* qkv = (const __bf16*)qkvp;
  __bf16* y = (__bf16*)yp;

  const int tid = threadIdx.x;
  const int lane = tid & 31, w = tid >> 5;
  const int n16 = lane & 15, hl = lane >> 4;
  const int qb = blockIdx.x;
  const int b = blockIdx.y >> 4, h = blockIdx.y & 15;
  const size_t baseBS = (size_t)b * S;
  const size_t RS = 3072;             // qkv row stride (elements)

  const int si = tid >> 1, sseg = tid & 1;  // staging: 2 threads per row

  // ---- stage Q block [128][64] via async DMA ----
  {
    const __bf16* gq =
        qkv + (baseBS + qb * 128 + si) * RS + h * 64 + sseg * 32;
    const __bf16* lq = Qs + si * 64 + sseg * 32;
#pragma unroll
    for (int c = 0; c < 4; ++c) async_copy16(gq + c * 8, lq + c * 8);
  }
  wait_async_0();
  __syncthreads();

  // Q fragments live in registers for the whole kernel (2 K-chunks of Ne=64).
  BF16Frag qf[2];
  {
    const __bf16* rp = Qs + (w * 16 + n16) * 64;
    load_fragA(qf[0], rp, 0, hl);
    load_fragA(qf[1], rp, 32, hl);
  }

  v8f acc_o[4];
#pragma unroll
  for (int e = 0; e < 4; ++e)
#pragma unroll
    for (int v = 0; v < 8; ++v) acc_o[e][v] = 0.0f;
  float m_r[8], l_r[8];
#pragma unroll
  for (int v = 0; v < 8; ++v) { m_r[v] = -__builtin_inff(); l_r[v] = 0.0f; }

  const int qrow_base = qb * 128 + w * 16;
  const float scale = 0.125f;  // 1/sqrt(64)

  for (int j = 0; j <= qb; ++j) {
    // ---- stage K block [128][64] (async) + transposed V [64][128] ----
    {
      const size_t rbase = (baseBS + j * 128 + si) * RS + h * 64 + sseg * 32;
      const __bf16* gk = qkv + rbase + 1024;
      const __bf16* lk = Ks + si * 64 + sseg * 32;
#pragma unroll
      for (int c = 0; c < 4; ++c) async_copy16(gk + c * 8, lk + c * 8);

      union { uint4 q[4]; __bf16 hv[32]; } u;
      const uint4* vs = (const uint4*)(qkv + rbase + 2048);
      u.q[0] = vs[0]; u.q[1] = vs[1]; u.q[2] = vs[2]; u.q[3] = vs[3];
#pragma unroll
      for (int e = 0; e < 32; ++e) Vts[(sseg * 32 + e) * 128 + si] = u.hv[e];
    }
    wait_async_0();
    __syncthreads();

    // ---- S = Q K^T : 8 tiles of 16x16 along kv ----
    v8f sacc[8];
#pragma unroll
    for (int nt = 0; nt < 8; ++nt) {
      v8f s = {0.f, 0.f, 0.f, 0.f, 0.f, 0.f, 0.f, 0.f};
#pragma unroll
      for (int c = 0; c < 2; ++c) {
        BF16Frag kf;
        load_fragB(kf, Ks + (nt * 16 + n16) * 64, c * 32, hl);
        s = wmma_bf16(qf[c].v, kf.v, s);
      }
      sacc[nt] = s;
    }

    // ---- scale + causal mask (lane holds col n16, rows v + 8*hl) ----
#pragma unroll
    for (int nt = 0; nt < 8; ++nt) {
      const int kvcol = j * 128 + nt * 16 + n16;
#pragma unroll
      for (int v = 0; v < 8; ++v) {
        const int qrow = qrow_base + v + 8 * hl;
        const float sv = sacc[nt][v] * scale;
        sacc[nt][v] = (kvcol <= qrow) ? sv : -__builtin_inff();
      }
    }

    // ---- online softmax ----
    float tmax[8];
#pragma unroll
    for (int v = 0; v < 8; ++v) {
      float t = sacc[0][v];
#pragma unroll
      for (int nt = 1; nt < 8; ++nt) t = fmaxf(t, sacc[nt][v]);
      t = fmaxf(t, __shfl_xor(t, 1, 32));
      t = fmaxf(t, __shfl_xor(t, 2, 32));
      t = fmaxf(t, __shfl_xor(t, 4, 32));
      t = fmaxf(t, __shfl_xor(t, 8, 32));
      tmax[v] = t;
    }
    float corr[8];
#pragma unroll
    for (int v = 0; v < 8; ++v) {
      const float mn = fmaxf(m_r[v], tmax[v]);
      corr[v] = __expf(m_r[v] - mn);
      m_r[v] = mn;
    }
    float rsum[8];
#pragma unroll
    for (int v = 0; v < 8; ++v) rsum[v] = 0.0f;
#pragma unroll
    for (int nt = 0; nt < 8; ++nt)
#pragma unroll
      for (int v = 0; v < 8; ++v) {
        const float p = __expf(sacc[nt][v] - m_r[v]);
        sacc[nt][v] = p;
        rsum[v] += p;
      }
#pragma unroll
    for (int v = 0; v < 8; ++v) {
      float r = rsum[v];
      r += __shfl_xor(r, 1, 32);
      r += __shfl_xor(r, 2, 32);
      r += __shfl_xor(r, 4, 32);
      r += __shfl_xor(r, 8, 32);
      l_r[v] = l_r[v] * corr[v] + r;
    }
#pragma unroll
    for (int e = 0; e < 4; ++e)
#pragma unroll
      for (int v = 0; v < 8; ++v) acc_o[e][v] *= corr[v];

    // ---- P (C/D layout) -> bf16 in per-wave LDS -> A fragments ----
    __bf16* Pw = Ps + w * (16 * 128);
#pragma unroll
    for (int nt = 0; nt < 8; ++nt)
#pragma unroll
      for (int v = 0; v < 8; ++v)
        Pw[(v + 8 * hl) * 128 + nt * 16 + n16] = (__bf16)sacc[nt][v];
    // same-wave DS ops are in-order; compiler inserts s_wait_dscnt for reuse

    // ---- O += P V : 4 kv chunks x 4 e-tiles ----
#pragma unroll
    for (int kc = 0; kc < 4; ++kc) {
      BF16Frag pf;
      load_fragA(pf, Pw + n16 * 128, kc * 32, hl);
#pragma unroll
      for (int e = 0; e < 4; ++e) {
        BF16Frag vf;
        load_fragB(vf, Vts + (e * 16 + n16) * 128, kc * 32, hl);
        acc_o[e] = wmma_bf16(pf.v, vf.v, acc_o[e]);
      }
    }
    __syncthreads();  // protect Ks/Vts before next staging
  }

  // ---- normalize and write y (bf16, head-interleaved [B*S][1024]) ----
#pragma unroll
  for (int v = 0; v < 8; ++v) {
    const float inv = 1.0f / l_r[v];
    const int row = qrow_base + v + 8 * hl;
#pragma unroll
    for (int e = 0; e < 4; ++e) {
      const int col = h * 64 + e * 16 + n16;
      y[(baseBS + row) * 1024 + col] = (__bf16)(acc_o[e][v] * inv);
    }
  }
}

// ---------------------------------------------------------------------------
// Host launcher
// ---------------------------------------------------------------------------
extern "C" void kernel_launch(void* const* d_in, const int* in_sizes, int n_in,
                              void* d_out, int out_size, void* d_ws,
                              size_t ws_size, hipStream_t stream) {
  (void)n_in; (void)out_size; (void)ws_size; (void)in_sizes;
  const float* x      = (const float*)d_in[0];  // [4,2048,1024]
  const float* W_attn = (const float*)d_in[1];  // [1024,3072]
  const float* b_attn = (const float*)d_in[2];  // [3072]
  const float* W_proj = (const float*)d_in[3];  // [1024,1024]
  const float* b_proj = (const float*)d_in[4];  // [1024]
  float* out = (float*)d_out;                   // [4,2048,1024] fp32

  const int B = 4, S = 2048, C = 1024, Nh = 16;
  const int M = B * S;  // 8192

  // workspace carve-up (bf16 buffers)
  char* ws = (char*)d_ws;
  size_t off = 0;
  void* qkv_bf = (void*)(ws + off); off += (size_t)M * 3 * C * 2;
  void* y_bf   = (void*)(ws + off); off += (size_t)M * C * 2;
  void* x_bf   = (void*)(ws + off); off += (size_t)M * C * 2;
  void* WattnT = (void*)(ws + off); off += (size_t)3 * C * C * 2;
  void* WprojT = (void*)(ws + off);

  // 0) convert x to bf16
  convert_f32_to_bf16<<<dim3((M * C) / 2048), 256, 0, stream>>>(
      x, x_bf, (long)M * C);

  // 1) weight transpose + bf16 convert
  transpose_to_bf16<<<dim3(3 * C / 32, C / 32), 256, 0, stream>>>(
      W_attn, WattnT, C, 3 * C);
  transpose_to_bf16<<<dim3(C / 32, C / 32), 256, 0, stream>>>(
      W_proj, WprojT, C, C);

  // 2) qkv = x @ W_attn + b_attn   (bf16 -> bf16)
  gemm_bf16_wmma<false><<<dim3(3 * C / 128, M / 128), 256, 0, stream>>>(
      x_bf, WattnT, b_attn, qkv_bf, M, 3 * C, C);

  // 3) flash attention -> y (bf16)
  attn_flash_wmma<<<dim3(S / 128, B * Nh), 256, 81920, stream>>>(
      qkv_bf, y_bf, S);

  // 4) out = y @ W_proj + b_proj   (bf16 -> fp32)
  gemm_bf16_wmma<true><<<dim3(C / 128, M / 128), 256, 0, stream>>>(
      y_bf, WprojT, b_proj, out, M, C, C);
}
